// AdvancedNet_70635032150274
// MI455X (gfx1250) — compile-verified, matrix operand order checked by reading, not run on hardware
//
#include <hip/hip_runtime.h>
#include <hip/hip_bf16.h>
#include <stdint.h>

// ---------------------------------------------------------------------------
// Sizes (fixed by the reference)
// ---------------------------------------------------------------------------
#define Bn    4
#define Tn    512
#define ROWS  (Bn * Tn)      // 2048 token rows
#define Hn    125            // lstm hidden per direction
#define G4    (4 * Hn)       // 500 gates
#define KP0   224            // 200 (WE+TE) padded to mult of 32
#define KP1   256            // 250 (2H)    padded to mult of 32
#define Mh    100            // mlp hidden
#define NPAD_G 512           // 500 -> 512 (32 n-tiles)
#define NPAD_M 112           // 100 -> 112 (7 n-tiles)

typedef __attribute__((ext_vector_type(16))) __bf16 v16bf;
typedef __attribute__((ext_vector_type(8)))  float  v8f;

union FragU { uint4 q[2]; v16bf v; };

// ---------------------------------------------------------------------------
// Fast transcendental helpers (single v_exp_f32 / v_rcp_f32 each)
// ---------------------------------------------------------------------------
__device__ __forceinline__ float fexp2(float x) { return __builtin_amdgcn_exp2f(x); }
__device__ __forceinline__ float frcp(float x)  { return __builtin_amdgcn_rcpf(x); }
__device__ __forceinline__ float fsigmoid(float x) {
    // 1 / (1 + 2^(-x*log2 e))
    return frcp(1.0f + fexp2(-1.44269504088896340736f * x));
}
__device__ __forceinline__ float ftanh(float x) {
    // tanh(x) = 1 - 2/(e^{2x}+1),  e^{2x} = 2^{x*2*log2 e}; inf/0 limits give +-1
    float e = fexp2(2.88539008177792681472f * x);
    return 1.0f - 2.0f * frcp(e + 1.0f);
}

__device__ __forceinline__ unsigned short f2bf(float f) {
    unsigned u = __float_as_uint(f);
    u += 0x7FFFu + ((u >> 16) & 1u);   // round-to-nearest-even
    return (unsigned short)(u >> 16);
}

// ---------------------------------------------------------------------------
// Embedding gather -> bf16 packed A matrix [ROWS, KP0] (zero K-padding)
// ---------------------------------------------------------------------------
__global__ void embed_pack(const int* __restrict__ widx, const int* __restrict__ tidx,
                           const float* __restrict__ wemb, const float* __restrict__ temb,
                           unsigned short* __restrict__ xbf) {
    int idx = blockIdx.x * 256 + threadIdx.x;
    if (idx >= ROWS * KP0) return;
    int row = idx / KP0, k = idx % KP0;
    float v = 0.0f;
    if (k < 100)      v = wemb[(size_t)widx[row] * 100 + k];
    else if (k < 200) v = temb[(size_t)tidx[row] * 100 + (k - 100)];
    xbf[idx] = f2bf(v);
}

// ---------------------------------------------------------------------------
// Generic f32 [R,C] -> bf16 [Rpad,Cpad] pad-convert (zeros in padding)
// ---------------------------------------------------------------------------
__global__ void convert_pad(const float* __restrict__ src, int R, int C,
                            unsigned short* __restrict__ dst, int Rpad, int Cpad) {
    int idx = blockIdx.x * 256 + threadIdx.x;
    if (idx >= Rpad * Cpad) return;
    int r = idx / Cpad, c = idx % Cpad;
    dst[idx] = (r < R && c < C) ? f2bf(src[(size_t)r * C + c]) : (unsigned short)0;
}

// ---------------------------------------------------------------------------
// bf16 WMMA GEMM: C[2048, Nld] = A[2048, KP] * W[Npad, KP]^T + bias
// One wave (block of 32) per 16x16 C-tile; K unrolled at compile time.
// A/B fragments follow the CDNA5 16-bit operand layout:
//   lanes 0-15 : row/col = lane,    K octets {kt*32+0..7, kt*32+16..23}
//   lanes 16-31: row/col = lane-16, K octets {kt*32+8..15, kt*32+24..31}
// ---------------------------------------------------------------------------
template <int KP>
__global__ void gemm_bf16(const unsigned short* __restrict__ A,
                          const unsigned short* __restrict__ W,
                          const float* __restrict__ bias,
                          float* __restrict__ C, int N, int Nld) {
    const int lane = threadIdx.x;
    const int l16  = lane & 15;
    const int hi   = lane >> 4;          // 0 or 1
    const int nb   = blockIdx.x * 16;
    const int mb   = blockIdx.y * 16;

    const unsigned short* arow = A + (size_t)(mb + l16) * KP;
    const unsigned short* wrow = W + (size_t)(nb + l16) * KP;

    v8f acc = {};
#pragma unroll
    for (int kt = 0; kt < KP / 32; ++kt) {
        const int k0 = kt * 32 + hi * 8;
        const int k1 = k0 + 16;
        FragU fa, fb;
        fa.q[0] = *(const uint4*)(arow + k0);
        fa.q[1] = *(const uint4*)(arow + k1);
        fb.q[0] = *(const uint4*)(wrow + k0);
        fb.q[1] = *(const uint4*)(wrow + k1);
        acc = __builtin_amdgcn_wmma_f32_16x16x32_bf16(
                  false, fa.v, false, fb.v, (short)0, acc, false, false);
    }

    const int n = nb + l16;
    if (n < N) {
        const float bv = bias[n];
#pragma unroll
        for (int r = 0; r < 8; ++r) {
            const int m = mb + hi * 8 + r;
            C[(size_t)m * Nld + n] = acc[r] + bv;
        }
    }
}

// ---------------------------------------------------------------------------
// Bidirectional LSTM layer: 2 blocks (dir = blockIdx.x), 512 threads each.
// Whh (500x125 f32 = 244 KB) lives in dynamic LDS; h double-buffered in LDS.
// Thread (b,u) owns cell (b, hidden-unit u): 4 gate dot-products of length 125
// per step, then the elementwise cell update. pre already contains x@Wih^T + b.
// Writes h into hout[b*T+t][colOff + u] (colOff = dir*H) -> concat layout.
// ---------------------------------------------------------------------------
#define SH_STRIDE 132   // per-batch h stride (odd-ish padding vs 128 banks)
__global__ void lstm_bidir(const float* __restrict__ pre_f, const float* __restrict__ pre_b,
                           const float* __restrict__ Whh_f, const float* __restrict__ Whh_b,
                           float* __restrict__ hout) {
    extern __shared__ float smem[];
    float* sW = smem;                    // 500*125 = 62500 floats
    float* sh = smem + G4 * Hn;          // 2 * 4 * SH_STRIDE floats

    const int dir = blockIdx.x;
    const float* pre = dir ? pre_b : pre_f;
    const float* Whh = dir ? Whh_b : Whh_f;
    const int colOff = dir * Hn;
    const int tid = threadIdx.x;

    for (int i = tid; i < G4 * Hn; i += 512) sW[i] = Whh[i];
    for (int i = tid; i < 2 * Bn * SH_STRIDE; i += 512) sh[i] = 0.0f;
    __syncthreads();

    const bool act = tid < Bn * Hn;      // 500 active threads
    const int b = tid / Hn;
    const int u = tid % Hn;
    const float* wi = sW + (size_t)(0 * Hn + u) * Hn;
    const float* wf = sW + (size_t)(1 * Hn + u) * Hn;
    const float* wg = sW + (size_t)(2 * Hn + u) * Hn;
    const float* wo = sW + (size_t)(3 * Hn + u) * Hn;

    float c = 0.0f;
    int buf = 0;
    for (int t = 0; t < Tn; ++t) {
        if (act) {
            const float* hprev = sh + buf * (Bn * SH_STRIDE) + b * SH_STRIDE;
            float ai = 0.f, af = 0.f, ag = 0.f, ao = 0.f;
#pragma unroll 5
            for (int k = 0; k < Hn; ++k) {
                const float hk = hprev[k];
                ai = fmaf(wi[k], hk, ai);
                af = fmaf(wf[k], hk, af);
                ag = fmaf(wg[k], hk, ag);
                ao = fmaf(wo[k], hk, ao);
            }
            const int te = dir ? (Tn - 1 - t) : t;     // reverse scan for dir=1
            const float* p = pre + ((size_t)b * Tn + te) * G4;
            const float gi = p[0 * Hn + u] + ai;
            const float gf = p[1 * Hn + u] + af;
            const float gg = p[2 * Hn + u] + ag;
            const float go = p[3 * Hn + u] + ao;
            c = fsigmoid(gf) * c + fsigmoid(gi) * ftanh(gg);
            const float h = fsigmoid(go) * ftanh(c);
            sh[(buf ^ 1) * (Bn * SH_STRIDE) + b * SH_STRIDE + u] = h;
            hout[((size_t)b * Tn + te) * (2 * Hn) + colOff + u] = h;
        }
        __syncthreads();
        buf ^= 1;
    }
}

// ---------------------------------------------------------------------------
// Pairwise biaffine score: out[b,i,j-1] = tanh(vh[b,i,:]+vm[b,j,:]).w + ob
// Block = 256 j's for a fixed (b,i); vm tile staged in LDS (stride 101: odd
// stride -> bank-conflict-free), vh row + w broadcast from LDS.
// ---------------------------------------------------------------------------
#define VM_STRIDE 101
__global__ void pairwise(const float* __restrict__ vh, const float* __restrict__ vm,
                         const float* __restrict__ w, const float* __restrict__ ob,
                         float* __restrict__ out) {
    extern __shared__ float smem[];
    float* svh = smem;                   // 100 (+pad)
    float* sw  = smem + 104;             // 100 (+pad)
    float* svm = smem + 208;             // 256 * VM_STRIDE

    const int tid = threadIdx.x;
    const int b  = blockIdx.z;
    const int i  = blockIdx.y;
    const int j0 = blockIdx.x * 256;

    if (tid < Mh) {
        svh[tid] = vh[((size_t)b * Tn + i) * Mh + tid];
        sw[tid]  = w[tid];
    }
    for (int idx = tid; idx < 256 * Mh; idx += 256) {
        const int jl = idx / Mh;
        const int m  = idx % Mh;
        svm[jl * VM_STRIDE + m] = vm[((size_t)b * Tn + j0 + jl) * Mh + m];
    }
    __syncthreads();

    const float* row = svm + tid * VM_STRIDE;
    float acc = 0.0f;
#pragma unroll 4
    for (int m = 0; m < Mh; ++m)
        acc += ftanh(svh[m] + row[m]) * sw[m];

    const int j = j0 + tid;
    if (j >= 1)
        out[((size_t)b * Tn + i) * (Tn - 1) + (j - 1)] = acc + ob[0];
}

// ---------------------------------------------------------------------------
// Host orchestration
// ---------------------------------------------------------------------------
extern "C" void kernel_launch(void* const* d_in, const int* in_sizes, int n_in,
                              void* d_out, int out_size, void* d_ws, size_t ws_size,
                              hipStream_t stream) {
    (void)in_sizes; (void)n_in; (void)out_size; (void)ws_size;

    const int*   widx    = (const int*)d_in[0];
    const int*   tidx    = (const int*)d_in[1];
    const float* wemb    = (const float*)d_in[2];
    const float* temb    = (const float*)d_in[3];
    const float* l0f_Wih = (const float*)d_in[4];
    const float* l0f_Whh = (const float*)d_in[5];
    const float* l0f_b   = (const float*)d_in[6];
    const float* l0b_Wih = (const float*)d_in[7];
    const float* l0b_Whh = (const float*)d_in[8];
    const float* l0b_b   = (const float*)d_in[9];
    const float* l1f_Wih = (const float*)d_in[10];
    const float* l1f_Whh = (const float*)d_in[11];
    const float* l1f_b   = (const float*)d_in[12];
    const float* l1b_Wih = (const float*)d_in[13];
    const float* l1b_Whh = (const float*)d_in[14];
    const float* l1b_b   = (const float*)d_in[15];
    const float* head_W  = (const float*)d_in[16];
    const float* head_b  = (const float*)d_in[17];
    const float* mod_W   = (const float*)d_in[18];
    const float* mod_b   = (const float*)d_in[19];
    const float* out_w   = (const float*)d_in[20];
    const float* out_b   = (const float*)d_in[21];

    char* ws = (char*)d_ws;
    size_t off = 0;
    auto alloc = [&](size_t bytes) -> void* {
        void* p = ws + off;
        off = (off + bytes + 255) & ~(size_t)255;
        return p;
    };

    unsigned short* xbf  = (unsigned short*)alloc((size_t)ROWS * KP0 * 2);
    unsigned short* w0f  = (unsigned short*)alloc((size_t)NPAD_G * KP0 * 2);
    unsigned short* w0b  = (unsigned short*)alloc((size_t)NPAD_G * KP0 * 2);
    unsigned short* w1f  = (unsigned short*)alloc((size_t)NPAD_G * KP1 * 2);
    unsigned short* w1b  = (unsigned short*)alloc((size_t)NPAD_G * KP1 * 2);
    unsigned short* whd  = (unsigned short*)alloc((size_t)NPAD_M * KP1 * 2);
    unsigned short* wmd  = (unsigned short*)alloc((size_t)NPAD_M * KP1 * 2);
    float* pre_f = (float*)alloc((size_t)ROWS * G4 * 4);   // reused: L0, L1, then vh
    float* pre_b = (float*)alloc((size_t)ROWS * G4 * 4);   // reused: L0, L1, then vm
    float* h0    = (float*)alloc((size_t)ROWS * 2 * Hn * 4);
    float* h1    = (float*)alloc((size_t)ROWS * 2 * Hn * 4);
    unsigned short* hbf = (unsigned short*)alloc((size_t)ROWS * KP1 * 2); // reused L0->L1
    float* vh = pre_f;
    float* vm = pre_b;

    auto cvt = [&](const float* src, int R, int C, unsigned short* dst, int Rpad, int Cpad) {
        int n = Rpad * Cpad;
        convert_pad<<<dim3((n + 255) / 256), dim3(256), 0, stream>>>(src, R, C, dst, Rpad, Cpad);
    };

    // Weight packs (bf16, zero-padded)
    cvt(l0f_Wih, G4, 200, w0f, NPAD_G, KP0);
    cvt(l0b_Wih, G4, 200, w0b, NPAD_G, KP0);
    cvt(l1f_Wih, G4, 250, w1f, NPAD_G, KP1);
    cvt(l1b_Wih, G4, 250, w1b, NPAD_G, KP1);
    cvt(head_W,  Mh, 250, whd, NPAD_M, KP1);
    cvt(mod_W,   Mh, 250, wmd, NPAD_M, KP1);

    // Embedding gather -> packed bf16 A
    {
        int n = ROWS * KP0;
        embed_pack<<<dim3((n + 255) / 256), dim3(256), 0, stream>>>(widx, tidx, wemb, temb, xbf);
    }

    const size_t lstm_smem = (size_t)(G4 * Hn + 2 * Bn * SH_STRIDE) * 4;

    // ---- Layer 0 ----
    gemm_bf16<KP0><<<dim3(NPAD_G / 16, ROWS / 16), dim3(32), 0, stream>>>(xbf, w0f, l0f_b, pre_f, G4, G4);
    gemm_bf16<KP0><<<dim3(NPAD_G / 16, ROWS / 16), dim3(32), 0, stream>>>(xbf, w0b, l0b_b, pre_b, G4, G4);
    lstm_bidir<<<dim3(2), dim3(512), lstm_smem, stream>>>(pre_f, pre_b, l0f_Whh, l0b_Whh, h0);
    cvt(h0, ROWS, 2 * Hn, hbf, ROWS, KP1);

    // ---- Layer 1 ----
    gemm_bf16<KP1><<<dim3(NPAD_G / 16, ROWS / 16), dim3(32), 0, stream>>>(hbf, w1f, l1f_b, pre_f, G4, G4);
    gemm_bf16<KP1><<<dim3(NPAD_G / 16, ROWS / 16), dim3(32), 0, stream>>>(hbf, w1b, l1b_b, pre_b, G4, G4);
    lstm_bidir<<<dim3(2), dim3(512), lstm_smem, stream>>>(pre_f, pre_b, l1f_Whh, l1b_Whh, h1);
    cvt(h1, ROWS, 2 * Hn, hbf, ROWS, KP1);

    // ---- Heads ----
    gemm_bf16<KP1><<<dim3(NPAD_M / 16, ROWS / 16), dim3(32), 0, stream>>>(hbf, whd, head_b, vh, Mh, Mh);
    gemm_bf16<KP1><<<dim3(NPAD_M / 16, ROWS / 16), dim3(32), 0, stream>>>(hbf, wmd, mod_b,  vm, Mh, Mh);

    // ---- Pairwise biaffine scores ----
    const size_t pw_smem = (size_t)(208 + 256 * VM_STRIDE) * 4;
    pairwise<<<dim3(Tn / 256, Tn, Bn), dim3(256), pw_smem, stream>>>(vh, vm, out_w, out_b, (float*)d_out);
}
